// LlamaAttention_63376537419810
// MI455X (gfx1250) — compile-verified, matrix operand order checked by reading, not run on hardware
//
#include <hip/hip_runtime.h>
#include <hip/hip_bf16.h>

// ---------------------------------------------------------------------------
// Types for CDNA5 WMMA (wave32): A/B = 16 x bf16 per lane, C/D = 8 x f32.
// ---------------------------------------------------------------------------
typedef __attribute__((ext_vector_type(16))) __bf16 bf16x16;
typedef __attribute__((ext_vector_type(8)))  __bf16 bf16x8;
typedef __attribute__((ext_vector_type(4)))  __bf16 bf16x4;
typedef __attribute__((ext_vector_type(2)))  __bf16 bf16x2;
typedef __attribute__((ext_vector_type(8)))  float  f32x8;
typedef __attribute__((ext_vector_type(4)))  float  f32x4;
typedef __attribute__((ext_vector_type(2)))  float  f32x2;
typedef __attribute__((ext_vector_type(4)))  unsigned u32x4;
typedef __attribute__((ext_vector_type(2)))  unsigned u32x2;

#define HIDDEN   4096
#define NHEADS   32
#define NKV      8
#define HDIM     128
#define SEQ      2048
#define BATCH    2
#define QKV_LD   6144   // 4096 q + 1024 k + 1024 v
#define KOFF     4096
#define VOFF     5120

// fp32 -> bf16 conversions expressed as vector converts so the backend can
// select the native packed cvt (v_cvt_pk_bf16_f32) when available.
__device__ __forceinline__ unsigned pack2(float a, float b) {
  f32x2 f; f[0] = a; f[1] = b;
  bf16x2 v = __builtin_convertvector(f, bf16x2);
  return __builtin_bit_cast(unsigned, v);
}

__device__ __forceinline__ u32x2 pack4(f32x4 f) {
  bf16x4 v = __builtin_convertvector(f, bf16x4);
  return __builtin_bit_cast(u32x2, v);
}

// Pack 8 consecutive floats (two f32x4) into 8 bf16 in a u32x4.
__device__ __forceinline__ u32x4 pack8(f32x4 lo, f32x4 hi) {
  u32x2 l = pack4(lo);
  u32x2 h = pack4(hi);
  u32x4 r;
  r[0] = l[0]; r[1] = l[1]; r[2] = h[0]; r[3] = h[1];
  return r;
}

// Load one 16x32 bf16 fragment (A row-major with k contiguous, or B stored
// N-major [n][k]) per the ISA 16-bit A-matrix layout:
//   lanes 0-15 : row = lane,    K = 0..7 (v0-3) and 16..23 (v4-7)
//   lanes16-31 : row = lane-16, K = 8..15        and 24..31
// Each lane: two contiguous 16-byte LDS loads.
__device__ __forceinline__ bf16x16 load_frag(const __bf16* base, int ld, int lane) {
  const int r  = lane & 15;
  const int kb = (lane >> 4) << 3;          // 0 or 8
  const __bf16* p = base + r * ld + kb;
  bf16x8 lo = *(const bf16x8*)(p);          // K = kb .. kb+7
  bf16x8 hi = *(const bf16x8*)(p + 16);     // K = kb+16 .. kb+23
  bf16x16 out;
#pragma unroll
  for (int i = 0; i < 8; ++i) { out[i] = lo[i]; out[i + 8] = hi[i]; }
  return out;
}

// ---------------------------------------------------------------------------
// C[M,N] = A[M,K] * B[K,N], fp32 in/out, bf16 WMMA compute.
// N, K compile-time so all strides fold into immediates.
// Block: 256 threads = 8 waves, 64x64 tile, BK=64 (4 WMMAs per barrier).
// Wave w: rows (w&3)*16, cols (w>>2)*32 .. +31  (two 16x16 accumulators).
// ---------------------------------------------------------------------------
template <int N, int K>
__global__ __launch_bounds__(256)
void gemm_bf16_wmma(const float* __restrict__ A, const float* __restrict__ B,
                    float* __restrict__ C) {
  __shared__ __align__(16) __bf16 As[64][64];   // [m][k]   8 KB
  __shared__ __align__(16) __bf16 Bs[64][64];   // [n][k]   8 KB (N-major)

  const int tid  = threadIdx.x;
  const int lane = tid & 31;
  const int wave = tid >> 5;
  const int bm   = blockIdx.y * 64;
  const int bn   = blockIdx.x * 64;
  const int wm   = (wave & 3) << 4;
  const int wn   = (wave >> 2) << 5;

  f32x8 acc0, acc1;
#pragma unroll
  for (int r = 0; r < 8; ++r) { acc0[r] = 0.f; acc1[r] = 0.f; }

  // A staging: 16 contiguous floats per thread -> 2 x b128 LDS stores
  const int arow = tid >> 2;             // 0..63
  const int acol = (tid & 3) << 4;       // 0,16,32,48
  // B staging: paired k rows, 8 n values per thread -> paired b32 LDS stores
  const int bk   = (tid & 31) << 1;      // 0,2,..,62
  const int bnb  = (tid >> 5) << 3;      // 0,8,..,56

  for (int k0 = 0; k0 < K; k0 += 64) {
    const float* ap  = A + (size_t)(bm + arow) * K + k0 + acol;
    const float* bp0 = B + (size_t)(k0 + bk) * N + bn + bnb;
    const float* bp1 = bp0 + N;
    __builtin_prefetch(ap + 64, 0, 1);   // next K-tile -> global_prefetch_b8

    {
      f32x4 a0 = *(const f32x4*)(ap);
      f32x4 a1 = *(const f32x4*)(ap + 4);
      f32x4 a2 = *(const f32x4*)(ap + 8);
      f32x4 a3 = *(const f32x4*)(ap + 12);
      *(u32x4*)&As[arow][acol]     = pack8(a0, a1);
      *(u32x4*)&As[arow][acol + 8] = pack8(a2, a3);
    }
    {
      f32x4 r0a = *(const f32x4*)(bp0);
      f32x4 r0b = *(const f32x4*)(bp0 + 4);
      f32x4 r1a = *(const f32x4*)(bp1);
      f32x4 r1b = *(const f32x4*)(bp1 + 4);
#pragma unroll
      for (int j = 0; j < 4; ++j) {
        *(unsigned*)&Bs[bnb + j][bk]     = pack2(r0a[j], r1a[j]);
        *(unsigned*)&Bs[bnb + 4 + j][bk] = pack2(r0b[j], r1b[j]);
      }
    }
    __syncthreads();

#pragma unroll
    for (int kk = 0; kk < 2; ++kk) {
      bf16x16 a  = load_frag(&As[wm][kk * 32],      64, lane);
      bf16x16 b0 = load_frag(&Bs[wn][kk * 32],      64, lane);
      bf16x16 b1 = load_frag(&Bs[wn + 16][kk * 32], 64, lane);
      acc0 = __builtin_amdgcn_wmma_f32_16x16x32_bf16(false, a, false, b0,
                                                     (short)0, acc0, false, false);
      acc1 = __builtin_amdgcn_wmma_f32_16x16x32_bf16(false, a, false, b1,
                                                     (short)0, acc1, false, false);
    }
    __syncthreads();
  }

  const int sub = (lane >> 4) << 3;      // C layout: v[r] is M=r / r+8
  const int col = lane & 15;
  float* cp = C + (size_t)(bm + wm + sub) * N + bn + wn + col;
#pragma unroll
  for (int r = 0; r < 8; ++r) {
    cp[(size_t)r * N]      = acc0[r];
    cp[(size_t)r * N + 16] = acc1[r];
  }
}

// ---------------------------------------------------------------------------
// NeoX RoPE applied in place to q (32 heads) and k (8 heads) of qkv buffer.
// One block per token; 40 heads * 64 pairs = 2560 work items.
// ---------------------------------------------------------------------------
__global__ __launch_bounds__(256)
void rope_kernel(const int* __restrict__ positions, float* __restrict__ qkv) {
  const int token = blockIdx.x;                 // 0 .. B*S-1
  const int s     = token & (SEQ - 1);
  float* tok = qkv + (size_t)token * QKV_LD;
  const float pos = (float)positions[s];
  for (int idx = threadIdx.x; idx < 40 * 64; idx += 256) {
    const int head = idx >> 6;
    const int i    = idx & 63;
    float* hp = (head < NHEADS) ? (tok + head * HDIM)
                                : (tok + KOFF + (head - NHEADS) * HDIM);
    // inv_freq = 10000^(-i/64) = exp(-i * ln(1e4)/64)
    const float fr = pos * __expf(-(float)i * 0.14391156f);
    const float cs = __cosf(fr), sn = __sinf(fr);
    const float x1 = hp[i], x2 = hp[i + 64];
    hp[i]      = x1 * cs - x2 * sn;
    hp[i + 64] = x2 * cs + x1 * sn;
  }
}

// ---------------------------------------------------------------------------
// Flash attention: one block per (b, head, 64-row q tile).
// Per 32-row K tile: S = Q*K^T via 8 waves * 4 WMMAs, fp32 online softmax,
// O = O*alpha + P*V via 4 WMMAs per wave. Causal tiles skipped entirely.
// ---------------------------------------------------------------------------
__global__ __launch_bounds__(256)
void attn_kernel(const float* __restrict__ qkv, float* __restrict__ out) {
  __shared__ __align__(16) __bf16 Qs[64][128];   // 16 KB
  __shared__ __align__(16) __bf16 Ks[32][128];   //  8 KB (N-major for QK^T)
  __shared__ __align__(16) __bf16 VTs[128][32];  //  8 KB (N-major for PV)
  __shared__ __align__(16) float  Ss[64][32];    //  8 KB
  __shared__ __align__(16) __bf16 Ps[64][32];    //  4 KB
  __shared__ float m_s[64], l_s[64], a_s[64];

  const int tid  = threadIdx.x;
  const int lane = tid & 31;
  const int wave = tid >> 5;
  const int q0   = blockIdx.x * 64;
  const int h    = blockIdx.y;
  const int b    = blockIdx.z;
  const int kvh  = h >> 2;                       // 32/8 = 4-way GQA
  const size_t tokbase = (size_t)(b * SEQ) * QKV_LD;

  // stage Q tile (64 x 128) once: 32 floats/thread -> 4 x b128 LDS stores
  {
    const int row = tid >> 2;
    const int cb  = (tid & 3) << 5;
    const float* src = qkv + tokbase + (size_t)(q0 + row) * QKV_LD + h * HDIM + cb;
#pragma unroll
    for (int g = 0; g < 4; ++g) {
      f32x4 lo = *(const f32x4*)(src + g * 8);
      f32x4 hi = *(const f32x4*)(src + g * 8 + 4);
      *(u32x4*)&Qs[row][cb + g * 8] = pack8(lo, hi);
    }
  }
  if (tid < 64) { m_s[tid] = -1e30f; l_s[tid] = 0.f; }

  f32x8 acc[4];
#pragma unroll
  for (int c = 0; c < 4; ++c)
#pragma unroll
    for (int r = 0; r < 8; ++r) acc[c][r] = 0.f;

  const int wm    = (wave >> 1) << 4;            // row group: 0/16/32/48
  const int wcol  = wave & 1;                    // column half for O
  const int numKT = (q0 >> 5) + 2;               // causal: kstart <= q0+63
  const float scale = 0.08838834764831845f;      // 128^-0.5

  __syncthreads();

  for (int kt = 0; kt < numKT; ++kt) {
    const int kstart = kt << 5;
    // stage K row-major (b128 stores) and V transposed (paired-row b32 stores)
    {
      const int krow = tid >> 3;                 // 0..31
      const int kcb  = (tid & 7) << 4;           // 0..112
      const float* ksrc = qkv + tokbase + (size_t)(kstart + krow) * QKV_LD
                          + KOFF + kvh * HDIM + kcb;
      f32x4 k0 = *(const f32x4*)(ksrc);
      f32x4 k1 = *(const f32x4*)(ksrc + 4);
      f32x4 k2 = *(const f32x4*)(ksrc + 8);
      f32x4 k3 = *(const f32x4*)(ksrc + 12);
      *(u32x4*)&Ks[krow][kcb]     = pack8(k0, k1);
      *(u32x4*)&Ks[krow][kcb + 8] = pack8(k2, k3);

      const int vr = (tid & 15) << 1;            // 0,2,..,30  (paired k rows)
      const int db = (tid >> 4) << 3;            // 0,8,..,120 (head-dim slice)
      const float* vsrc0 = qkv + tokbase + (size_t)(kstart + vr) * QKV_LD
                           + VOFF + kvh * HDIM + db;
      const float* vsrc1 = vsrc0 + QKV_LD;
      f32x4 v0a = *(const f32x4*)(vsrc0);
      f32x4 v0b = *(const f32x4*)(vsrc0 + 4);
      f32x4 v1a = *(const f32x4*)(vsrc1);
      f32x4 v1b = *(const f32x4*)(vsrc1 + 4);
#pragma unroll
      for (int j = 0; j < 4; ++j) {
        *(unsigned*)&VTs[db + j][vr]     = pack2(v0a[j], v1a[j]);
        *(unsigned*)&VTs[db + 4 + j][vr] = pack2(v0b[j], v1b[j]);
      }
    }
    __syncthreads();

    // S tile: each wave one 16x16 of the 64x32 score block
    {
      const int sn = (wave & 1) << 4;
      f32x8 sc;
#pragma unroll
      for (int r = 0; r < 8; ++r) sc[r] = 0.f;
#pragma unroll
      for (int kk = 0; kk < 4; ++kk) {           // head dim 128 = 4 x 32
        bf16x16 a = load_frag(&Qs[wm][kk * 32], 128, lane);
        bf16x16 k = load_frag(&Ks[sn][kk * 32], 128, lane);
        sc = __builtin_amdgcn_wmma_f32_16x16x32_bf16(false, a, false, k,
                                                     (short)0, sc, false, false);
      }
      const int sub = (lane >> 4) << 3;
      const int n   = lane & 15;
#pragma unroll
      for (int r = 0; r < 8; ++r) {
        const int row  = wm + r + sub;
        const int colg = kstart + sn + n;
        float v = sc[r] * scale;
        if (colg > q0 + row) v = -1e30f;         // causal mask
        Ss[row][sn + n] = v;
      }
    }
    __syncthreads();

    // online softmax, one thread per row; P written as packed b128
    if (tid < 64) {
      float mo = m_s[tid];
      float mx = mo;
#pragma unroll
      for (int j = 0; j < 32; ++j) mx = fmaxf(mx, Ss[tid][j]);
      const float alpha = __expf(mo - mx);
      float sum = 0.f;
      float p[32];
#pragma unroll
      for (int j = 0; j < 32; ++j) {
        p[j] = __expf(Ss[tid][j] - mx);
        sum += p[j];
      }
#pragma unroll
      for (int g = 0; g < 4; ++g) {
        u32x4 w;
        w[0] = pack2(p[g * 8 + 0], p[g * 8 + 1]);
        w[1] = pack2(p[g * 8 + 2], p[g * 8 + 3]);
        w[2] = pack2(p[g * 8 + 4], p[g * 8 + 5]);
        w[3] = pack2(p[g * 8 + 6], p[g * 8 + 7]);
        *(u32x4*)&Ps[tid][g * 8] = w;
      }
      m_s[tid] = mx;
      l_s[tid] = l_s[tid] * alpha + sum;
      a_s[tid] = alpha;
    }
    __syncthreads();

    // O = O*alpha + P*V   (wave covers rows wm..wm+15, cols wcol*64..+63)
    {
      bf16x16 a = load_frag(&Ps[wm][0], 32, lane);
      const int sub = (lane >> 4) << 3;
#pragma unroll
      for (int c = 0; c < 4; ++c) {
#pragma unroll
        for (int r = 0; r < 8; ++r) acc[c][r] *= a_s[wm + r + sub];
        bf16x16 v = load_frag(&VTs[wcol * 64 + c * 16][0], 32, lane);
        acc[c] = __builtin_amdgcn_wmma_f32_16x16x32_bf16(false, a, false, v,
                                                         (short)0, acc[c], false, false);
      }
    }
    __syncthreads();
  }

  // normalize and write [B*S][4096] (heads concatenated)
  {
    const int sub = (lane >> 4) << 3;
    const int n   = lane & 15;
    float* op = out + (size_t)(b * SEQ + q0 + wm + sub) * HIDDEN
                + h * HDIM + wcol * 64 + n;
#pragma unroll
    for (int c = 0; c < 4; ++c)
#pragma unroll
      for (int r = 0; r < 8; ++r) {
        const int row = wm + r + sub;
        op[(size_t)r * HIDDEN + c * 16] = acc[c][r] / l_s[row];
      }
  }
}

// ---------------------------------------------------------------------------
// Launch: qkv GEMM -> RoPE -> flash attention -> output GEMM
// ---------------------------------------------------------------------------
extern "C" void kernel_launch(void* const* d_in, const int* in_sizes, int n_in,
                              void* d_out, int out_size, void* d_ws, size_t ws_size,
                              hipStream_t stream) {
  (void)in_sizes; (void)n_in; (void)out_size; (void)ws_size;
  const int*   positions = (const int*)  d_in[0];
  const float* hidden    = (const float*)d_in[1];
  const float* w_qkv     = (const float*)d_in[2];
  const float* w_o       = (const float*)d_in[3];
  float*       out       = (float*)d_out;

  float* qkv  = (float*)d_ws;                               // [4096][6144]
  float* attn = qkv + (size_t)(BATCH * SEQ) * QKV_LD;       // [4096][4096]

  const int M = BATCH * SEQ;                                // 4096 tokens

  // qkv = hidden @ w_qkv    (4096 x 4096 x 6144)
  gemm_bf16_wmma<QKV_LD, HIDDEN><<<dim3(QKV_LD / 64, M / 64), 256, 0, stream>>>(
      hidden, w_qkv, qkv);

  // RoPE in place on q and k
  rope_kernel<<<dim3(M), 256, 0, stream>>>(positions, qkv);

  // flash attention -> attn [4096][4096]
  attn_kernel<<<dim3(SEQ / 64, NHEADS, BATCH), 256, 0, stream>>>(qkv, attn);

  // out = attn @ w_o        (4096 x 4096 x 4096)
  gemm_bf16_wmma<HIDDEN, HIDDEN><<<dim3(HIDDEN / 64, M / 64), 256, 0, stream>>>(
      attn, w_o, out);
}
